// MultiheadCrossAttention_91233695302570
// MI455X (gfx1250) — matrix operand order repaired and checked by reading, not compiled, not run on hardware
//
#include <hip/hip_runtime.h>

#define BDIM   4
#define SEQLQ  2048
#define SEQLK  2048
#define DMODEL 512
#define NHEAD  8
#define HDIM   64
#define QKSCALE 0.044194173824159216f   // 1/sqrt(512)

typedef __bf16 bf16;
typedef __attribute__((ext_vector_type(16))) __bf16 v16bf;
typedef __attribute__((ext_vector_type(8)))  __bf16 v8bf;
typedef __attribute__((ext_vector_type(8)))  float  v8f;

static __device__ __forceinline__ bf16 f2bf(float f) {
  unsigned int u = __builtin_bit_cast(unsigned int, f);
  u += 0x7FFFu + ((u >> 16) & 1u);               // round to nearest even
  unsigned short h = (unsigned short)(u >> 16);
  return __builtin_bit_cast(bf16, h);
}

static __device__ __forceinline__ v8f wmma_bf16(v16bf a, v16bf b, v8f c) {
  return __builtin_amdgcn_wmma_f32_16x16x32_bf16(false, a, false, b, (short)0, c,
                                                 false, false);
}

// A fragment (16x32, bf16): lane holds row = lane%16; K chunks [b..b+7] and
// [b+16..b+23] with b = (lane/16)*8 (per CDNA5 ISA 16-bit A layout).
static __device__ __forceinline__ v16bf load_a_frag(const bf16* base, int ld) {
  int lane = threadIdx.x & 31;
  const bf16* p = base + (lane & 15) * ld + ((lane >> 4) << 3);
  v8bf lo = *(const v8bf*)p;
  v8bf hi = *(const v8bf*)(p + 16);
  v16bf f;
#pragma unroll
  for (int i = 0; i < 8; ++i) { f[i] = lo[i]; f[i + 8] = hi[i]; }
  return f;
}

// B fragment (32x16, bf16) where element (k,n) = base[n*ld + k]:
// lane holds column n = lane%16; K chunks [b..b+7] and [b+16..b+23], b = (lane/16)*8.
static __device__ __forceinline__ v16bf load_b_frag(const bf16* base, int ld) {
  int lane = threadIdx.x & 31;
  const bf16* p = base + (lane & 15) * ld + ((lane >> 4) << 3);
  v8bf lo = *(const v8bf*)p;
  v8bf hi = *(const v8bf*)(p + 16);
  v16bf f;
#pragma unroll
  for (int i = 0; i < 8; ++i) { f[i] = lo[i]; f[i + 8] = hi[i]; }
  return f;
}

// Async global->LDS copy of 16 bytes per lane (gfx1250 ASYNCcnt path).
// ldsoff = low 32 bits of the generic LDS address (== LDS byte offset).
static __device__ __forceinline__ void async_copy_b128(const void* gaddr,
                                                       void* lds_generic) {
  unsigned int ldsoff = (unsigned int)(unsigned long long)lds_generic;
  unsigned long long ga = (unsigned long long)gaddr;
  asm volatile("global_load_async_to_lds_b128 %0, %1, off"
               :
               : "v"(ldsoff), "v"(ga)
               : "memory");
}
static __device__ __forceinline__ void async_wait0() {
  asm volatile("s_wait_asynccnt 0x0" ::: "memory");
}

__global__ void __launch_bounds__(256) cvt_f32_bf16(const float* __restrict__ in,
                                                    bf16* __restrict__ out, int n) {
  int i = (blockIdx.x * 256 + threadIdx.x) * 4;
  if (i + 3 < n) {
    float4 v = *(const float4*)(in + i);
    out[i]     = f2bf(v.x);
    out[i + 1] = f2bf(v.y);
    out[i + 2] = f2bf(v.z);
    out[i + 3] = f2bf(v.w);
  }
}

// C = A(MxK) * W^T  where W is (N,K) row-major, so B(k,n) = W[n*K + k].
// Each wave computes a 32x64 tile (2 A-frags x 4 B-frags = 8 WMMAs/k-step).
// k-loop unrolled by 2 with ping-pong fragment buffers: one buffer computes
// while the other buffer's loads are in flight; no register rotation copies.
// Requires Kd % 64 == 0.
// mode 0: bf16 store. mode 1: f32 store with bias + residual epilogue.
__global__ void __launch_bounds__(256) gemm_bf16(const bf16* __restrict__ A,
                                                 const bf16* __restrict__ W,
                                                 bf16* __restrict__ outB,
                                                 float* __restrict__ outF,
                                                 const float* __restrict__ bias,
                                                 const float* __restrict__ resid,
                                                 int M, int N, int Kd, int mode) {
  int wave = blockIdx.x * 8 + (threadIdx.x >> 5);
  int lane = threadIdx.x & 31;
  int ncols = N >> 6;
  int mtile = wave / ncols;
  int nbase = (wave % ncols) << 6;
  if (mtile * 32 >= M) return;

  const bf16* A0 = A + (size_t)(mtile * 32) * Kd;
  const bf16* A1 = A0 + (size_t)16 * Kd;
  const bf16* W0 = W + (size_t)nbase * Kd;
  v8f acc[8] = {};

  // buffer A: k-step kk ; buffer B: k-step kk+32
  v16bf a0A = load_a_frag(A0, Kd);
  v16bf a1A = load_a_frag(A1, Kd);
  v16bf bA[4];
#pragma unroll
  for (int j = 0; j < 4; ++j)
    bA[j] = load_b_frag(W0 + (size_t)(j * 16) * Kd, Kd);

  for (int kk = 0; kk < Kd; kk += 64) {
    int k1 = kk + 32;                       // always < Kd (Kd % 64 == 0)
    v16bf a0B = load_a_frag(A0 + k1, Kd);
    v16bf a1B = load_a_frag(A1 + k1, Kd);
    v16bf bB[4];
#pragma unroll
    for (int j = 0; j < 4; ++j)
      bB[j] = load_b_frag(W0 + (size_t)(j * 16) * Kd + k1, Kd);

#pragma unroll
    for (int j = 0; j < 4; ++j) {
      acc[j]     = wmma_bf16(a0A, bA[j], acc[j]);
      acc[4 + j] = wmma_bf16(a1A, bA[j], acc[4 + j]);
    }

    int k2 = (kk + 64 < Kd) ? kk + 64 : 0;  // last-iter prefetch is discarded
    a0A = load_a_frag(A0 + k2, Kd);
    a1A = load_a_frag(A1 + k2, Kd);
#pragma unroll
    for (int j = 0; j < 4; ++j)
      bA[j] = load_b_frag(W0 + (size_t)(j * 16) * Kd + k2, Kd);

#pragma unroll
    for (int j = 0; j < 4; ++j) {
      acc[j]     = wmma_bf16(a0B, bB[j], acc[j]);
      acc[4 + j] = wmma_bf16(a1B, bB[j], acc[4 + j]);
    }
  }

#pragma unroll
  for (int mi = 0; mi < 2; ++mi) {
#pragma unroll
    for (int j = 0; j < 4; ++j) {
#pragma unroll
      for (int r = 0; r < 8; ++r) {
        int row = mtile * 32 + mi * 16 + r + ((lane >> 4) << 3);
        int col = nbase + j * 16 + (lane & 15);
        float v = acc[mi * 4 + j][r];
        if (mode) {
          v += bias[col] + resid[(size_t)row * N + col];
          outF[(size_t)row * N + col] = v;
        } else {
          outB[(size_t)row * N + col] = f2bf(v);
        }
      }
    }
  }
}

// Flash attention: block = 8 waves sharing (b,h); each wave owns 16 q-rows.
// LK processed in 32-column tiles; K tile staged via async global->LDS copy,
// V tile transposed into LDS manually.
__global__ void __launch_bounds__(256) flash_attn(const bf16* __restrict__ Q,
                                                  const bf16* __restrict__ K,
                                                  const bf16* __restrict__ V,
                                                  const float* __restrict__ mask,
                                                  bf16* __restrict__ O) {
  __shared__ bf16 Ktile[32 * 64];       // [kcol][d]
  __shared__ bf16 VT[64 * 32];          // [d][kcol]
  __shared__ bf16 Pws[8][16 * 32];      // per-wave P transpose scratch

  int qblk = blockIdx.x & 15;
  int bh   = blockIdx.x >> 4;
  int h = bh & 7, b = bh >> 3;
  int wave = threadIdx.x >> 5, lane = threadIdx.x & 31;
  int qbase = qblk * 128 + wave * 16;

  const bf16* Qb = Q + ((size_t)(b * SEQLQ + qbase)) * DMODEL + h * HDIM;
  const bf16* Kb = K + ((size_t)b * SEQLK) * DMODEL + h * HDIM;
  const bf16* Vb = V + ((size_t)b * SEQLK) * DMODEL + h * HDIM;
  const float* mrow = mask + (size_t)b * SEQLK;

  v16bf qa0 = load_a_frag(Qb, DMODEL);        // d = 0..31
  v16bf qa1 = load_a_frag(Qb + 32, DMODEL);   // d = 32..63

  v8f acc[4] = {};
  float mst[8], lst[8];
#pragma unroll
  for (int r = 0; r < 8; ++r) { mst[r] = -1e30f; lst[r] = 0.f; }

  int stg_n = threadIdx.x >> 3;             // k-row 0..31
  int stg_d = (threadIdx.x & 7) << 3;       // d chunk

  for (int kt = 0; kt < SEQLK; kt += 32) {
    __syncthreads();
    {
      const bf16* gk = Kb + (size_t)(kt + stg_n) * DMODEL + stg_d;
      async_copy_b128(gk, &Ktile[stg_n * 64 + stg_d]);   // ASYNCcnt path
      v8bf vv = *(const v8bf*)(Vb + (size_t)(kt + stg_n) * DMODEL + stg_d);
#pragma unroll
      for (int i = 0; i < 8; ++i) VT[(stg_d + i) * 32 + stg_n] = vv[i];
      async_wait0();                                     // K tile landed in LDS
    }
    __syncthreads();

    if (kt + 32 < SEQLK) {                               // prefetch next tiles
      __builtin_prefetch(Kb + (size_t)(kt + 32 + stg_n) * DMODEL + stg_d, 0, 3);
      __builtin_prefetch(Vb + (size_t)(kt + 32 + stg_n) * DMODEL + stg_d, 0, 3);
    }

    // S(16x32) = Q(16x64) . K_tile^T : B(d,n) = Ktile[n*64 + d]
    v8f s[2] = {};
#pragma unroll
    for (int j = 0; j < 2; ++j) {
      s[j] = wmma_bf16(qa0, load_b_frag(&Ktile[(j * 16) * 64], 64), s[j]);
      s[j] = wmma_bf16(qa1, load_b_frag(&Ktile[(j * 16) * 64 + 32], 64), s[j]);
    }

    // scale + attention-mask additive term (column-indexed)
#pragma unroll
    for (int j = 0; j < 2; ++j) {
      float madd = (1.0f - mrow[kt + j * 16 + (lane & 15)]) * -100000.0f;
#pragma unroll
      for (int r = 0; r < 8; ++r) s[j][r] = s[j][r] * QKSCALE + madd;
    }

    // online softmax; row r maps to q-row (r + 8*(lane/16)) => reduce within
    // each 16-lane half via xor shuffles.
    float mnew[8], corr[8];
#pragma unroll
    for (int r = 0; r < 8; ++r) {
      float mx = fmaxf(s[0][r], s[1][r]);
#pragma unroll
      for (int d = 1; d <= 8; d <<= 1) mx = fmaxf(mx, __shfl_xor(mx, d, 32));
      mnew[r] = fmaxf(mst[r], mx);
      corr[r] = __expf(mst[r] - mnew[r]);
      mst[r]  = mnew[r];
    }
    v8f p0, p1;
#pragma unroll
    for (int r = 0; r < 8; ++r) {
      p0[r] = __expf(s[0][r] - mnew[r]);
      p1[r] = __expf(s[1][r] - mnew[r]);
      float sm = p0[r] + p1[r];
#pragma unroll
      for (int d = 1; d <= 8; d <<= 1) sm += __shfl_xor(sm, d, 32);
      lst[r] = lst[r] * corr[r] + sm;
    }
#pragma unroll
    for (int j = 0; j < 4; ++j)
#pragma unroll
      for (int r = 0; r < 8; ++r) acc[j][r] *= corr[r];

    // P: C-layout -> A-layout via per-wave LDS scratch (same-wave DS is in-order)
    bf16* pw = &Pws[wave][0];
    int prow = (lane >> 4) << 3;
    int pcol = lane & 15;
#pragma unroll
    for (int r = 0; r < 8; ++r) {
      pw[(r + prow) * 32 + pcol]      = f2bf(p0[r]);
      pw[(r + prow) * 32 + 16 + pcol] = f2bf(p1[r]);
    }
    v16bf pa = load_a_frag(pw, 32);

    // O(16x64) += P(16x32) . V_tile(32x64) : B(k,n) = VT[n*32 + k]
#pragma unroll
    for (int j = 0; j < 4; ++j)
      acc[j] = wmma_bf16(pa, load_b_frag(&VT[(j * 16) * 32], 32), acc[j]);
  }

  bf16* Ob = O + ((size_t)(b * SEQLQ + qbase)) * DMODEL + h * HDIM;
#pragma unroll
  for (int j = 0; j < 4; ++j)
#pragma unroll
    for (int r = 0; r < 8; ++r) {
      int row = r + ((lane >> 4) << 3);
      int col = j * 16 + (lane & 15);
      Ob[(size_t)row * DMODEL + col] = f2bf(acc[j][r] / lst[r]);
    }
}

extern "C" void kernel_launch(void* const* d_in, const int* in_sizes, int n_in,
                              void* d_out, int out_size, void* d_ws, size_t ws_size,
                              hipStream_t stream) {
  const float* x  = (const float*)d_in[0];
  const float* y  = (const float*)d_in[1];
  const float* am = (const float*)d_in[2];
  const float* Wq = (const float*)d_in[3];
  const float* Wk = (const float*)d_in[4];
  const float* Wv = (const float*)d_in[5];
  const float* Wo = (const float*)d_in[6];
  const float* bo = (const float*)d_in[7];
  float* out = (float*)d_out;

  const size_t NX = (size_t)BDIM * SEQLQ * DMODEL;   // 4,194,304
  const size_t NW = (size_t)DMODEL * DMODEL;         // 262,144

  bf16* p  = (bf16*)d_ws;
  bf16* xb  = p; p += NX;
  bf16* yb  = p; p += NX;
  bf16* Wqb = p; p += NW;
  bf16* Wkb = p; p += NW;
  bf16* Wvb = p; p += NW;
  bf16* Wob = p; p += NW;
  bf16* Qb  = p; p += NX;
  bf16* Kb  = p; p += NX;
  bf16* Vb  = p; p += NX;
  bf16* Ob  = p; p += NX;

  cvt_f32_bf16<<<(unsigned)((NX / 4 + 255) / 256), 256, 0, stream>>>(x, xb, (int)NX);
  cvt_f32_bf16<<<(unsigned)((NX / 4 + 255) / 256), 256, 0, stream>>>(y, yb, (int)NX);
  cvt_f32_bf16<<<(unsigned)((NW / 4 + 255) / 256), 256, 0, stream>>>(Wq, Wqb, (int)NW);
  cvt_f32_bf16<<<(unsigned)((NW / 4 + 255) / 256), 256, 0, stream>>>(Wk, Wkb, (int)NW);
  cvt_f32_bf16<<<(unsigned)((NW / 4 + 255) / 256), 256, 0, stream>>>(Wv, Wvb, (int)NW);
  cvt_f32_bf16<<<(unsigned)((NW / 4 + 255) / 256), 256, 0, stream>>>(Wo, Wob, (int)NW);

  const int M = BDIM * SEQLQ, N = DMODEL, Kd = DMODEL;
  const int gblocks = (M / 32) * (N / 64) / 8;   // 256 blocks of 8 waves
  gemm_bf16<<<gblocks, 256, 0, stream>>>(xb, Wqb, Qb, nullptr, nullptr, nullptr, M, N, Kd, 0);
  gemm_bf16<<<gblocks, 256, 0, stream>>>(yb, Wkb, Kb, nullptr, nullptr, nullptr, M, N, Kd, 0);
  gemm_bf16<<<gblocks, 256, 0, stream>>>(yb, Wvb, Vb, nullptr, nullptr, nullptr, M, N, Kd, 0);

  flash_attn<<<BDIM * NHEAD * (SEQLQ / 128), 256, 0, stream>>>(Qb, Kb, Vb, am, Ob);

  gemm_bf16<<<gblocks, 256, 0, stream>>>(Ob, Wob, nullptr, out, bo, x, M, N, Kd, 1);
}
